// AFMLayer_59047210385708
// MI455X (gfx1250) — matrix operand order repaired and checked by reading, not compiled, not run on hardware
//
#include <hip/hip_runtime.h>
#include <hip/hip_bf16.h>
#include <math.h>

typedef __attribute__((ext_vector_type(16))) _Float16 v16h;
typedef __attribute__((ext_vector_type(8)))  float    v8f;

#define FDIM 40
#define EDIM 64
#define ADIM 64
#define NPAIR 780              // F*(F-1)/2
#define NPAIR_PAD 784          // padded to 16
#define NTILE 49               // 784 / 16
#define NTHREADS 256
#define XPAD 4                 // row stride 68 floats = 272B: 16B-aligned, bank-shifted

__global__ __launch_bounds__(NTHREADS)
void afm_fused_kernel(const float* __restrict__ x,   // (B, F, E)
                      const float* __restrict__ W,   // (E, A)
                      const float* __restrict__ bvec,// (A)
                      const float* __restrict__ hvec,// (A)
                      const float* __restrict__ pw,  // (E)
                      float* __restrict__ out)       // (B)
{
    __shared__ __align__(16) float xs[FDIM][EDIM + XPAD];
    __shared__ _Float16  Ws[EDIM][ADIM + 2];
    __shared__ float     bs[ADIM], hs[ADIM], pws[EDIM];
    __shared__ float     score[NPAIR_PAD];       // scores -> exp -> weights
    __shared__ unsigned char iiS[NPAIR_PAD], jjS[NPAIR_PAD];
    __shared__ float     red[8];
    __shared__ float     pooledPart[4][EDIM];

    const int tid = threadIdx.x;
    const int bId = blockIdx.x;

    // ---- stage x (B,F,E) row to LDS via float4 (coalesced b128) ----
    {
        const float4* xg = (const float4*)(x + (size_t)bId * FDIM * EDIM);
        for (int v = tid; v < FDIM * EDIM / 4; v += NTHREADS) {
            const int f = (4 * v) / EDIM;
            const int c = (4 * v) % EDIM;
            *(float4*)&xs[f][c] = xg[v];
        }
    }
    for (int idx = tid; idx < EDIM * ADIM; idx += NTHREADS)
        Ws[idx / ADIM][idx % ADIM] = (_Float16)W[idx];
    if (tid < ADIM) { bs[tid] = bvec[tid]; hs[tid] = hvec[tid]; }
    if (tid < EDIM) pws[tid] = pw[tid];
    // pair index tables (np.triu_indices order); padding points at row 0
    for (int p = tid; p < NPAIR_PAD; p += NTHREADS) {
        if (p < NPAIR) {
            int i = 0, rem = p;
            while (rem >= FDIM - 1 - i) { rem -= FDIM - 1 - i; ++i; }
            iiS[p] = (unsigned char)i;
            jjS[p] = (unsigned char)(i + 1 + rem);
        } else { iiS[p] = 0; jjS[p] = 0; }
    }
    __syncthreads();

    const int wave = tid >> 5;
    const int lane = tid & 31;
    const int half = lane >> 4;   // 0: lanes 0-15, 1: lanes 16-31
    const int lm   = lane & 15;

    // ---- preload all B-fragments of W: [n_subtile][k_chunk], resident in VGPRs ----
    // B k order used here (contiguous, same as A): lanes 0-15 hold K=0..15 of col N=lane,
    //                           lanes 16-31 hold K=16..31 of col N=lane-16.
    v16h bfrag[4][2];
#pragma unroll
    for (int ns = 0; ns < 4; ++ns) {
#pragma unroll
        for (int kc = 0; kc < 2; ++kc) {
            const int n  = ns * 16 + lm;
            const int kb = kc * 32 + half * 16;
#pragma unroll
            for (int j = 0; j < 16; ++j)
                bfrag[ns][kc][j] = Ws[kb + j][n];
        }
    }

    // ---- main loop: 16-pair M-tiles, GEMM via WMMA, fused bias+relu+dot(h) ----
    for (int tile = wave; tile < NTILE; tile += 8) {
        const int p  = tile * 16 + lm;
        const int i1 = iiS[p];
        const int j1 = jjS[p];
        const float4* ri = (const float4*)&xs[i1][0];
        const float4* rj = (const float4*)&xs[j1][0];

        // A k order used here (contiguous, same as B): lane<16 -> row M=lane, K = 0..15;
        //                           lane>=16 -> row M=lane-16, K = 16..31.
        // Padded rows (p>=780) read row 0 twice: harmless, their outputs are never read.
        v16h afrag[2];
#pragma unroll
        for (int kc = 0; kc < 2; ++kc) {
#pragma unroll
            for (int c = 0; c < 2; ++c) {            // chunk base K: +0 and +8
                const int q = (kc * 32 + half * 16 + c * 8) >> 2;  // float4 index
                const float4 a0 = ri[q],     b0 = rj[q];
                const float4 a1 = ri[q + 1], b1 = rj[q + 1];
                afrag[kc][c * 8 + 0] = (_Float16)(a0.x * b0.x);
                afrag[kc][c * 8 + 1] = (_Float16)(a0.y * b0.y);
                afrag[kc][c * 8 + 2] = (_Float16)(a0.z * b0.z);
                afrag[kc][c * 8 + 3] = (_Float16)(a0.w * b0.w);
                afrag[kc][c * 8 + 4] = (_Float16)(a1.x * b1.x);
                afrag[kc][c * 8 + 5] = (_Float16)(a1.y * b1.y);
                afrag[kc][c * 8 + 6] = (_Float16)(a1.z * b1.z);
                afrag[kc][c * 8 + 7] = (_Float16)(a1.w * b1.w);
            }
        }

        // 4 independent accumulator chains: all k-step-0 WMMAs, then all k-step-1
        v8f c[4];
#pragma unroll
        for (int ns = 0; ns < 4; ++ns) {
            v8f z = {};
            c[ns] = __builtin_amdgcn_wmma_f32_16x16x32_f16(false, afrag[0],
                                                           false, bfrag[ns][0],
                                                           (short)0, z, false, false);
        }
#pragma unroll
        for (int ns = 0; ns < 4; ++ns) {
            c[ns] = __builtin_amdgcn_wmma_f32_16x16x32_f16(false, afrag[1],
                                                           false, bfrag[ns][1],
                                                           (short)0, c[ns], false, false);
        }

        float part[8];
#pragma unroll
        for (int r = 0; r < 8; ++r) part[r] = 0.0f;
#pragma unroll
        for (int ns = 0; ns < 4; ++ns) {
            // C layout: lane -> N = ns*16 + lm ; VGPR r -> M = r + 8*half
            const int   n  = ns * 16 + lm;
            const float bn = bs[n];
            const float hn = hs[n];
#pragma unroll
            for (int r = 0; r < 8; ++r) {
                float v = c[ns][r] + bn;
                v = v > 0.0f ? v : 0.0f;      // relu
                part[r] += v * hn;            // dot with h (partial over N)
            }
        }
        // reduce partial scores across the 16 lanes of each half
#pragma unroll
        for (int r = 0; r < 8; ++r) {
            float v = part[r];
            v += __shfl_xor(v, 1, 32);
            v += __shfl_xor(v, 2, 32);
            v += __shfl_xor(v, 4, 32);
            v += __shfl_xor(v, 8, 32);
            part[r] = v;
        }
        if (lm == 0) {
#pragma unroll
            for (int r = 0; r < 8; ++r)
                score[tile * 16 + half * 8 + r] = part[r];
        }
    }
    __syncthreads();

    // ---- softmax over 780 real pairs, scaled by num_pairs ----
    float lmax = -1e30f;
    for (int p = tid; p < NPAIR; p += NTHREADS) lmax = fmaxf(lmax, score[p]);
#pragma unroll
    for (int d = 16; d >= 1; d >>= 1) lmax = fmaxf(lmax, __shfl_xor(lmax, d, 32));
    if (lane == 0) red[wave] = lmax;
    __syncthreads();
    if (tid == 0) {
        float m = red[0];
        for (int i = 1; i < 8; ++i) m = fmaxf(m, red[i]);
        red[0] = m;
    }
    __syncthreads();
    const float gmax = red[0];
    __syncthreads();

    float lsum = 0.0f;
    for (int p = tid; p < NPAIR; p += NTHREADS) {
        const float e = __expf(score[p] - gmax);
        score[p] = e;
        lsum += e;
    }
#pragma unroll
    for (int d = 16; d >= 1; d >>= 1) lsum += __shfl_xor(lsum, d, 32);
    if (lane == 0) red[wave] = lsum;
    __syncthreads();
    if (tid == 0) {
        float s = 0.0f;
        for (int i = 0; i < 8; ++i) s += red[i];
        red[0] = s;
    }
    __syncthreads();
    const float scale = (float)NPAIR / red[0];
    for (int p = tid; p < NPAIR; p += NTHREADS) score[p] *= scale;   // now w[p]
    __syncthreads();

    // ---- pooled[e] = sum_p w[p]*bi[p][e] (bi recomputed from LDS x), out = pooled . p_w ----
    {
        const int e = tid & 63;
        const int g = tid >> 6;           // 4 groups strided over pairs
        float acc = 0.0f;
        for (int p = g; p < NPAIR; p += 4)
            acc += score[p] * xs[iiS[p]][e] * xs[jjS[p]][e];
        pooledPart[g][e] = acc;
    }
    __syncthreads();
    if (tid < EDIM) {
        const float pooled = pooledPart[0][tid] + pooledPart[1][tid]
                           + pooledPart[2][tid] + pooledPart[3][tid];
        pooledPart[0][tid] = pooled * pws[tid];
    }
    __syncthreads();
    if (tid == 0) {
        float s = 0.0f;
        for (int i = 0; i < EDIM; ++i) s += pooledPart[0][i];
        out[bId] = s;
    }
}

extern "C" void kernel_launch(void* const* d_in, const int* in_sizes, int n_in,
                              void* d_out, int out_size, void* d_ws, size_t ws_size,
                              hipStream_t stream) {
    const float* x  = (const float*)d_in[0];   // (B, 40, 64)
    const float* W  = (const float*)d_in[1];   // (64, 64)
    const float* b  = (const float*)d_in[2];   // (64)
    const float* h  = (const float*)d_in[3];   // (64)
    const float* pw = (const float*)d_in[4];   // (64)
    float* out = (float*)d_out;                // (B, 1)

    const int B = in_sizes[0] / (FDIM * EDIM); // 2048
    afm_fused_kernel<<<B, NTHREADS, 0, stream>>>(x, W, b, h, pw, out);
}
